// CVAE_38268158607907
// MI455X (gfx1250) — matrix operand order repaired and checked by reading, not compiled, not run on hardware
//
#include <hip/hip_runtime.h>
#include <math.h>

// ---------------------------------------------------------------------------
// CVAE graph encoder/decoder for MI455X (gfx1250, wave32, WMMA 16x16x32 f16)
// Round 2: unconditional vectorized loads, LDS-staged transposed weights.
// ---------------------------------------------------------------------------

typedef __attribute__((ext_vector_type(16))) _Float16 v16h;
typedef __attribute__((ext_vector_type(8)))  _Float16 v8h;
typedef __attribute__((ext_vector_type(4)))  _Float16 v4h;
typedef __attribute__((ext_vector_type(8)))  float    v8f;

#define NN 100000    // nodes
#define NE 3200000   // edges
#define DF 256       // node feature dim
#define GH 20        // GC hidden
#define DH 40        // decoder hidden
#define ZD 32        // latent dim
#define ED 256       // expr dim
#define TSTRIDE 32   // padded column stride for GC activations
#define KH 288       // padded K for the mu/ls GEMM (20 + 256 -> 288)

// Build an A fragment (contiguous k order, same as the B fragments) from 16 f32 values at p[ka..ka+15],
// scaled by s.  All loads aligned float4.
__device__ __forceinline__ v16h a_frag_from_f32(const float* __restrict__ p,
                                                int ka, float s) {
  float4 p0 = *(const float4*)(p + ka);
  float4 p1 = *(const float4*)(p + ka + 4);
  float4 p2 = *(const float4*)(p + ka + 8);
  float4 p3 = *(const float4*)(p + ka + 12);
  v16h a;
  a[0]  = (_Float16)(p0.x * s); a[1]  = (_Float16)(p0.y * s);
  a[2]  = (_Float16)(p0.z * s); a[3]  = (_Float16)(p0.w * s);
  a[4]  = (_Float16)(p1.x * s); a[5]  = (_Float16)(p1.y * s);
  a[6]  = (_Float16)(p1.z * s); a[7]  = (_Float16)(p1.w * s);
  a[8]  = (_Float16)(p2.x * s); a[9]  = (_Float16)(p2.y * s);
  a[10] = (_Float16)(p2.z * s); a[11] = (_Float16)(p2.w * s);
  a[12] = (_Float16)(p3.x * s); a[13] = (_Float16)(p3.y * s);
  a[14] = (_Float16)(p3.z * s); a[15] = (_Float16)(p3.w * s);
  return a;
}

// A fragment from contiguous f16 storage (two aligned 8-half vectors).
__device__ __forceinline__ v16h a_frag_from_f16(const _Float16* __restrict__ p,
                                                int ka) {
  v8h lo = *(const v8h*)(p + ka);
  v8h hi = *(const v8h*)(p + ka + 8);
  v16h a;
#pragma unroll
  for (int i = 0; i < 8; ++i) { a[i] = lo[i]; a[8 + i] = hi[i]; }
  return a;
}

// ---------------------------------------------------------------------------
__global__ void k_zero(float* __restrict__ p, int n) {
  int i = blockIdx.x * blockDim.x + threadIdx.x;
  if (i < n) p[i] = 0.0f;
}

__global__ void k_degrees(const int* __restrict__ snd, const int* __restrict__ rcv,
                          float* __restrict__ deg_s, float* __restrict__ deg_r, int E) {
  int e = blockIdx.x * blockDim.x + threadIdx.x;
  if (e < E) {
    atomicAdd(&deg_s[snd[e]], 1.0f);
    atomicAdd(&deg_r[rcv[e]], 1.0f);
  }
}

// ---------------------------------------------------------------------------
// T[row, 0..19] = softmax(relu(X@W + b)) * rsqrt(max(deg_s,1))
// X must have >= KPAD valid columns per row (zeros beyond K).
// If USE_DEG_R: X rows are scaled by rsqrt(max(deg_r,1)) during A-load.
// W^T is staged zero-padded into LDS ([col][k] f16), so B fragments are two
// unconditional ds_load_b128 each.  One wave per 16-row tile.
// ---------------------------------------------------------------------------
template <int K, int KPAD, int LDA, bool USE_DEG_R>
__global__ void k_gc_transform(const float* __restrict__ X,
                               const float* __restrict__ W,     // [K x 20]
                               const float* __restrict__ bias,  // [20]
                               const float* __restrict__ deg_s,
                               const float* __restrict__ deg_r,
                               float* __restrict__ T) {
  __shared__ alignas(32) _Float16 ldsWt[32 * KPAD];  // [col][k], zero padded
  for (int idx = threadIdx.x; idx < 32 * KPAD; idx += blockDim.x) {
    int col = idx / KPAD, k = idx % KPAD;
    float v = (col < GH && k < K) ? W[k * GH + col] : 0.0f;
    ldsWt[idx] = (_Float16)v;
  }
  __syncthreads();

  const int lane = threadIdx.x & 31;
  const int wave = threadIdx.x >> 5;
  const int tile_row = (blockIdx.x * (blockDim.x >> 5) + wave) * 16;

  const int m  = lane & 15;
  const int hh = (lane >> 4) & 1;
  const int n  = lane & 15;
  const int row  = tile_row + m;
  const int rowc = (row < NN) ? row : (NN - 1);  // clamp: loads unconditional
  const int ka = hh ? 16 : 0;
  const int kb = hh ? 16 : 0;

  float ascale = 1.0f;
  if (USE_DEG_R) ascale = rsqrtf(fmaxf(deg_r[rowc], 1.0f));
  const float* Xrow = X + (size_t)rowc * LDA;

  v8f acc0 = {}; v8f acc1 = {};
#pragma unroll
  for (int k0 = 0; k0 < KPAD; k0 += 32) {
    v16h a  = a_frag_from_f32(Xrow + k0, ka, ascale);
    v16h b0 = *(const v16h*)&ldsWt[(size_t)n * KPAD + k0 + kb];
    v16h b1 = *(const v16h*)&ldsWt[(size_t)(16 + n) * KPAD + k0 + kb];
    acc0 = __builtin_amdgcn_wmma_f32_16x16x32_f16(false, a, false, b0, (short)0, acc0, false, false);
    acc1 = __builtin_amdgcn_wmma_f32_16x16x32_f16(false, a, false, b1, (short)0, acc1, false, false);
  }

  // Epilogue: bias + relu + row softmax over 20 valid cols + deg_s scaling.
  const bool  c1v  = (16 + n) < GH;
  const float bia0 = bias[n];
  const float bia1 = bias[c1v ? (16 + n) : (GH - 1)];  // clamped load
#pragma unroll
  for (int r = 0; r < 8; ++r) {
    int rowm = tile_row + r + 8 * hh;
    int rowmc = (rowm < NN) ? rowm : (NN - 1);
    float v0 = fmaxf(acc0[r] + bia0, 0.0f);
    float v1 = c1v ? fmaxf(acc1[r] + bia1, 0.0f) : -3.0e38f;
    float mx = fmaxf(v0, v1);
#pragma unroll
    for (int off = 8; off; off >>= 1) mx = fmaxf(mx, __shfl_xor(mx, off, 32));
    float e0 = __expf(v0 - mx);
    float e1 = c1v ? __expf(v1 - mx) : 0.0f;
    float sm = e0 + e1;
#pragma unroll
    for (int off = 8; off; off >>= 1) sm += __shfl_xor(sm, off, 32);
    float sc = rsqrtf(fmaxf(deg_s[rowmc], 1.0f)) / sm;
    if (rowm < NN) {
      T[(size_t)rowm * TSTRIDE + n] = e0 * sc;
      if (c1v) T[(size_t)rowm * TSTRIDE + 16 + n] = e1 * sc;
    }
  }
}

// ---------------------------------------------------------------------------
// AGG[rcv[e]] += T[snd[e]]   (segment_sum over edges; 20 f32 atomics / edge)
// ---------------------------------------------------------------------------
__global__ void k_aggregate(const float* __restrict__ T, const int* __restrict__ snd,
                            const int* __restrict__ rcv, float* __restrict__ AGG, int E) {
  int e = blockIdx.x * blockDim.x + threadIdx.x;
  if (e >= E) return;
  const float4* src = (const float4*)(T + (size_t)snd[e] * TSTRIDE);
  float* dst = AGG + (size_t)rcv[e] * TSTRIDE;
  float4 v0 = src[0], v1 = src[1], v2 = src[2], v3 = src[3], v4 = src[4];
  atomicAdd(dst + 0, v0.x);  atomicAdd(dst + 1, v0.y);
  atomicAdd(dst + 2, v0.z);  atomicAdd(dst + 3, v0.w);
  atomicAdd(dst + 4, v1.x);  atomicAdd(dst + 5, v1.y);
  atomicAdd(dst + 6, v1.z);  atomicAdd(dst + 7, v1.w);
  atomicAdd(dst + 8, v2.x);  atomicAdd(dst + 9, v2.y);
  atomicAdd(dst + 10, v2.z); atomicAdd(dst + 11, v2.w);
  atomicAdd(dst + 12, v3.x); atomicAdd(dst + 13, v3.y);
  atomicAdd(dst + 14, v3.z); atomicAdd(dst + 15, v3.w);
  atomicAdd(dst + 16, v4.x); atomicAdd(dst + 17, v4.y);
  atomicAdd(dst + 18, v4.z); atomicAdd(dst + 19, v4.w);
}

// ---------------------------------------------------------------------------
// Hh[row, 0..287] = f16([AGG*rsqrt(max(deg_r,1)) (20), nodes (256), zeros (12)])
// One thread per 4-column group; all loads/stores aligned.
// ---------------------------------------------------------------------------
__global__ void k_prep_h(const float* __restrict__ AGG, const float* __restrict__ nodes,
                         const float* __restrict__ deg_r, _Float16* __restrict__ Hh) {
  int idx = blockIdx.x * blockDim.x + threadIdx.x;
  if (idx >= NN * (KH / 4)) return;
  int row = idx / (KH / 4);
  int g   = idx % (KH / 4);
  float4 v = make_float4(0.0f, 0.0f, 0.0f, 0.0f);
  if (g < 5) {
    v = *(const float4*)(AGG + (size_t)row * TSTRIDE + 4 * g);
    float rs = rsqrtf(fmaxf(deg_r[row], 1.0f));
    v.x *= rs; v.y *= rs; v.z *= rs; v.w *= rs;
  } else if (g < 69) {
    v = *(const float4*)(nodes + (size_t)row * DF + (4 * g - GH));
  }
  v4h h;
  h[0] = (_Float16)v.x; h[1] = (_Float16)v.y;
  h[2] = (_Float16)v.z; h[3] = (_Float16)v.w;
  *(v4h*)(Hh + (size_t)row * KH + 4 * g) = h;
}

// ---------------------------------------------------------------------------
// mu = H@Wmu + bmu ; ls = H@Wls + bls ; z = mu + (1e-4 + exp(0.5 ls))*eps
// H is the staged f16 [N x 288] buffer.  [Wmu | Wls]^T staged in LDS (64 cols).
// Column tiles 0..1 -> mu, tiles 2..3 -> ls.
// ---------------------------------------------------------------------------
__global__ void k_mu_ls_z(const _Float16* __restrict__ Hh, const float* __restrict__ eps,
                          const float* __restrict__ Wmu, const float* __restrict__ bmu,
                          const float* __restrict__ Wls, const float* __restrict__ bls,
                          float* __restrict__ out_mu, float* __restrict__ out_ls,
                          _Float16* __restrict__ Zh) {
  __shared__ alignas(32) _Float16 ldsWt[64 * KH];  // 36 KB: [col][k]
  for (int idx = threadIdx.x; idx < 64 * KH; idx += blockDim.x) {
    int col = idx / KH, k = idx % KH;
    float v = 0.0f;
    if (k < GH + DF) v = (col < ZD) ? Wmu[(size_t)k * ZD + col]
                                    : Wls[(size_t)k * ZD + (col - ZD)];
    ldsWt[idx] = (_Float16)v;
  }
  __syncthreads();

  const int lane = threadIdx.x & 31;
  const int wave = threadIdx.x >> 5;
  const int tile_row = (blockIdx.x * (blockDim.x >> 5) + wave) * 16;
  const int m  = lane & 15;
  const int hh = (lane >> 4) & 1;
  const int n  = lane & 15;
  const int row  = tile_row + m;
  const int rowc = (row < NN) ? row : (NN - 1);
  const int ka = hh ? 16 : 0;
  const int kb = hh ? 16 : 0;
  const _Float16* Hrow = Hh + (size_t)rowc * KH;

  v8f acc[4];
#pragma unroll
  for (int t = 0; t < 4; ++t) acc[t] = (v8f){};

#pragma unroll
  for (int k0 = 0; k0 < KH; k0 += 32) {
    v16h a = a_frag_from_f16(Hrow + k0, ka);
#pragma unroll
    for (int t = 0; t < 4; ++t) {
      v16h b = *(const v16h*)&ldsWt[(size_t)(t * 16 + n) * KH + k0 + kb];
      acc[t] = __builtin_amdgcn_wmma_f32_16x16x32_f16(false, a, false, b, (short)0, acc[t], false, false);
    }
  }

#pragma unroll
  for (int t = 0; t < 2; ++t) {
    int c = t * 16 + n;
    float bm = bmu[c], bl = bls[c];
#pragma unroll
    for (int r = 0; r < 8; ++r) {
      int rowm  = tile_row + r + 8 * hh;
      int rowmc = (rowm < NN) ? rowm : (NN - 1);
      float mu = acc[t][r] + bm;
      float ls = acc[t + 2][r] + bl;
      float ep = eps[(size_t)rowmc * ZD + c];  // clamped, unconditional
      float z  = mu + (1.0e-4f + __expf(0.5f * ls)) * ep;
      if (rowm < NN) {
        out_mu[(size_t)rowm * ZD + c] = mu;
        out_ls[(size_t)rowm * ZD + c] = ls;
        Zh[(size_t)rowm * ZD + c] = (_Float16)z;
      }
    }
  }
}

// ---------------------------------------------------------------------------
// d = relu(z @ Wd1 + bd1)  [16x40, staged in LDS as 16x64 f16]
// X = d @ Wd2 + bd2        [16x256]
// Wd1^T (48x32) and Wd2^T (256x64) staged zero-padded in LDS.
// ---------------------------------------------------------------------------
__global__ void k_decoder(const _Float16* __restrict__ Zh,
                          const float* __restrict__ Wd1, const float* __restrict__ bd1,
                          const float* __restrict__ Wd2, const float* __restrict__ bd2,
                          float* __restrict__ Xout) {
  __shared__ alignas(32) _Float16 ldsW1t[48 * 32];    // 3 KB  [col][k]
  __shared__ alignas(32) _Float16 ldsW2t[256 * 64];   // 32 KB [col][k]
  __shared__ alignas(32) _Float16 dstage[8 * 16 * 64];// 16 KB, per-wave 16x64 tile

  for (int idx = threadIdx.x; idx < 48 * 32; idx += blockDim.x) {
    int col = idx / 32, k = idx % 32;
    ldsW1t[idx] = (_Float16)((col < DH) ? Wd1[(size_t)k * DH + col] : 0.0f);
  }
  for (int idx = threadIdx.x; idx < 256 * 64; idx += blockDim.x) {
    int col = idx / 64, k = idx % 64;
    ldsW2t[idx] = (_Float16)((k < DH) ? Wd2[(size_t)k * ED + col] : 0.0f);
  }
  __syncthreads();

  const int lane = threadIdx.x & 31;
  const int wave = threadIdx.x >> 5;
  const int tile_row = (blockIdx.x * (blockDim.x >> 5) + wave) * 16;
  const int m  = lane & 15;
  const int hh = (lane >> 4) & 1;
  const int n  = lane & 15;
  const int row  = tile_row + m;
  const int rowc = (row < NN) ? row : (NN - 1);
  const int ka = hh ? 16 : 0;
  const int kb = hh ? 16 : 0;
  _Float16* my = &dstage[wave * 16 * 64];

  // --- stage 1: d = relu(z @ Wd1 + bd1), K = 32 exact, 3 col tiles ---
  v16h a = a_frag_from_f16(Zh + (size_t)rowc * ZD, ka);
  v8f accd[3];
#pragma unroll
  for (int t = 0; t < 3; ++t) {
    v16h b = *(const v16h*)&ldsW1t[(size_t)(t * 16 + n) * 32 + kb];
    v8f cz = {};
    accd[t] = __builtin_amdgcn_wmma_f32_16x16x32_f16(false, a, false, b, (short)0, cz, false, false);
  }
#pragma unroll
  for (int t = 0; t < 3; ++t) {
    int cg = t * 16 + n;
    float bv = bd1[(cg < DH) ? cg : (DH - 1)];
#pragma unroll
    for (int r = 0; r < 8; ++r) {
      int rl = r + 8 * hh;
      float dv = (cg < DH) ? fmaxf(accd[t][r] + bv, 0.0f) : 0.0f;
      my[rl * 64 + cg] = (_Float16)dv;
    }
  }
#pragma unroll
  for (int r = 0; r < 8; ++r) my[(r + 8 * hh) * 64 + 48 + n] = (_Float16)0.0f;
  __syncthreads();

  // --- stage 2: X = d @ Wd2 + bd2; K = 64 padded, 16 col tiles of 16 ---
  v16h a2[2];
#pragma unroll
  for (int kk = 0; kk < 2; ++kk) a2[kk] = a_frag_from_f16(my + m * 64 + kk * 32, ka);

#pragma unroll
  for (int t = 0; t < 16; ++t) {
    int cg = t * 16 + n;
    v8f acc = {};
#pragma unroll
    for (int kk = 0; kk < 2; ++kk) {
      v16h b = *(const v16h*)&ldsW2t[(size_t)cg * 64 + kk * 32 + kb];
      acc = __builtin_amdgcn_wmma_f32_16x16x32_f16(false, a2[kk], false, b, (short)0, acc, false, false);
    }
    float bv = bd2[cg];
#pragma unroll
    for (int r = 0; r < 8; ++r) {
      int rowm = tile_row + r + 8 * hh;
      if (rowm < NN) Xout[(size_t)rowm * ED + cg] = acc[r] + bv;
    }
  }
}

// ---------------------------------------------------------------------------
extern "C" void kernel_launch(void* const* d_in, const int* in_sizes, int n_in,
                              void* d_out, int out_size, void* d_ws, size_t ws_size,
                              hipStream_t stream) {
  (void)in_sizes; (void)n_in; (void)out_size; (void)ws_size;

  const float* nodes     = (const float*)d_in[0];
  const int*   senders   = (const int*)d_in[1];
  const int*   receivers = (const int*)d_in[2];
  const float* eps       = (const float*)d_in[3];
  const float* W_gc1     = (const float*)d_in[4];
  const float* b_gc1     = (const float*)d_in[5];
  const float* W_gc2     = (const float*)d_in[6];
  const float* b_gc2     = (const float*)d_in[7];
  const float* W_mu      = (const float*)d_in[8];
  const float* b_mu      = (const float*)d_in[9];
  const float* W_ls      = (const float*)d_in[10];
  const float* b_ls      = (const float*)d_in[11];
  const float* W_d1      = (const float*)d_in[12];
  const float* b_d1      = (const float*)d_in[13];
  const float* W_d2      = (const float*)d_in[14];
  const float* b_d2      = (const float*)d_in[15];

  float* Xout   = (float*)d_out;                 // [N,256]
  float* mu_out = Xout + (size_t)NN * ED;        // [N,32]
  float* ls_out = mu_out + (size_t)NN * ZD;      // [N,32]

  // workspace carve (~92 MB total)
  char* w = (char*)d_ws;
  size_t off = 0;
  auto carve = [&](size_t bytes) -> void* {
    void* p = w + off;
    off += (bytes + 255) & ~(size_t)255;
    return p;
  };
  float*    deg_s = (float*)carve((size_t)NN * sizeof(float));
  float*    deg_r = (float*)carve((size_t)NN * sizeof(float));
  float*    T     = (float*)carve((size_t)NN * TSTRIDE * sizeof(float));
  float*    AGG   = (float*)carve((size_t)NN * TSTRIDE * sizeof(float));
  _Float16* Hh    = (_Float16*)carve((size_t)NN * KH * sizeof(_Float16));
  _Float16* Zh    = (_Float16*)carve((size_t)NN * ZD * sizeof(_Float16));

  const dim3 blk(256);
  const int tiles   = (NN + 15) / 16;
  const int tblocks = (tiles + 7) / 8;   // 8 waves (tiles) per block

  // degrees
  k_zero<<<(NN + 255) / 256, blk, 0, stream>>>(deg_s, NN);
  k_zero<<<(NN + 255) / 256, blk, 0, stream>>>(deg_r, NN);
  k_degrees<<<(NE + 255) / 256, blk, 0, stream>>>(senders, receivers, deg_s, deg_r, NE);

  // GC layer 1: transform(nodes) -> T ; aggregate -> AGG
  k_zero<<<(NN * TSTRIDE + 255) / 256, blk, 0, stream>>>(AGG, NN * TSTRIDE);
  k_gc_transform<DF, DF, DF, false><<<tblocks, blk, 0, stream>>>(
      nodes, W_gc1, b_gc1, deg_s, deg_r, T);
  k_aggregate<<<(NE + 255) / 256, blk, 0, stream>>>(T, senders, receivers, AGG, NE);

  // GC layer 2: deg_r scaling of layer-1 output folded into A-load
  k_gc_transform<GH, TSTRIDE, TSTRIDE, true><<<tblocks, blk, 0, stream>>>(
      AGG, W_gc2, b_gc2, deg_s, deg_r, T);
  k_zero<<<(NN * TSTRIDE + 255) / 256, blk, 0, stream>>>(AGG, NN * TSTRIDE);
  k_aggregate<<<(NE + 255) / 256, blk, 0, stream>>>(T, senders, receivers, AGG, NE);

  // concat + cast: Hh = f16([AGG*rsqrt(deg_r), nodes, 0])
  k_prep_h<<<(NN * (KH / 4) + 255) / 256, blk, 0, stream>>>(AGG, nodes, deg_r, Hh);

  // mu / logsig2 / z, then decoder
  k_mu_ls_z<<<tblocks, blk, 0, stream>>>(Hh, eps, W_mu, b_mu, W_ls, b_ls,
                                         mu_out, ls_out, Zh);
  k_decoder<<<tblocks, blk, 0, stream>>>(Zh, W_d1, b_d1, W_d2, b_d2, Xout);
}
